// LASERHiddenExtractorELMo_52596169507213
// MI455X (gfx1250) — matrix operand order repaired and checked by reading, not compiled, not run on hardware
//
#include <hip/hip_runtime.h>

#define S_LEN 128
#define BATCH 256
#define EMB   320
#define HID   512
#define GATES 2048   // 4*HID
#define BH    (BATCH * HID)
#define BROW  40     // LDS B-tile row stride in bf16 (32 data + 8 pad -> 80B, bank-conflict free)
#define BTILE (64 * BROW)   // one B tile in halves

typedef __attribute__((ext_vector_type(16))) __bf16 v16bf;
typedef __attribute__((ext_vector_type(8)))  __bf16 v8bf;
typedef __attribute__((ext_vector_type(8)))  float  v8f;
typedef __attribute__((ext_vector_type(4)))  int    v4i;

// CDNA5 async global->LDS copy path (ASYNCcnt), with a safe synchronous fallback.
#if defined(__has_builtin)
#if __has_builtin(__builtin_amdgcn_global_load_async_to_lds_b128) && \
    __has_builtin(__builtin_amdgcn_s_wait_asynccnt)
#define ASYNC_LDS 1
#endif
#endif
#ifndef ASYNC_LDS
#define ASYNC_LDS 0
#endif

#if ASYNC_LDS
typedef __attribute__((address_space(1))) v4i* gptr_b128;   // global 16B chunk
typedef __attribute__((address_space(3))) v4i* lptr_b128;   // LDS 16B chunk
#endif

__device__ __forceinline__ void stage16(__bf16* ldst, const __bf16* gsrc) {
#if ASYNC_LDS
    __builtin_amdgcn_global_load_async_to_lds_b128(
        (gptr_b128)(void*)gsrc, (lptr_b128)(void*)ldst, 0, 0);
#else
    *(v8bf*)ldst = *(const v8bf*)gsrc;
#endif
}

__device__ __forceinline__ void stage_wait() {
#if ASYNC_LDS
    __builtin_amdgcn_s_wait_asynccnt(0);
#endif
}

__device__ __forceinline__ __bf16 to_bf16(float f) {
    unsigned u = __builtin_bit_cast(unsigned, f);
    u += 0x7FFFu + ((u >> 16) & 1u);               // round-to-nearest-even
    unsigned short s = (unsigned short)(u >> 16);
    return __builtin_bit_cast(__bf16, s);
}

__device__ __forceinline__ float sigmoid_(float x) {
    return 1.0f / (1.0f + __expf(-x));
}

// acc[0..3] (one wave's 16x64 strip of C) += A[16 rows, K] * W[64 cols, K]^T.
// A: row-major [*, lda] bf16, read direct from global (rows are private per wave).
// W: row-major [2048, K] bf16; the 64x32 tile for the whole WG is staged in LDS
// (256 threads x 16B async DMA), double-buffered one K-chunk ahead so the
// ASYNCcnt transfer overlaps the WMMA stream of the previous chunk.
__device__ __forceinline__ void mma_panel_lds(
    const __bf16* __restrict__ A, int lda, int K,
    const __bf16* __restrict__ W, int n0,
    int tid, int lm, int lh, int mrow,
    __bf16* bsm, v8f acc[4])
{
    const int nl  = tid >> 2;   // 0..63: local output column staged by this thread
    const int seg = tid & 3;    // 16-byte segment within the 32-half K chunk
    const __bf16* arow = A + (size_t)mrow * lda;
    const __bf16* wsrc = W + (size_t)(n0 + nl) * K + seg * 8;
    __bf16* ldst0 = bsm + nl * BROW + seg * 8;
    __bf16* ldst1 = ldst0 + BTILE;

    __syncthreads();                 // previous panel's LDS reads fully drained
    stage16(ldst0, wsrc);            // prologue: chunk 0 -> buf 0
    const int nch = K >> 5;
    for (int i = 0; i < nch; ++i) {
        const int k = i << 5;
        const __bf16* cbuf = (i & 1) ? (bsm + BTILE) : bsm;
        stage_wait();                // my chunk-i fragment has landed in LDS
        __syncthreads();             // whole tile visible; buf[i^1] reads (iter i-1) drained
        if (i + 1 < nch)
            stage16((i & 1) ? ldst0 : ldst1, wsrc + k + 32);  // prefetch chunk i+1
        // 16-bit A 16x32 fragment: lanes 0-15 hold K {k..k+7, k+16..k+23},
        // lanes 16-31 hold K {k+8..k+15, k+24..k+31}; M = lane%16.
        v8bf alo = *(const v8bf*)(arow + k + lh * 8);
        v8bf ahi = *(const v8bf*)(arow + k + 16 + lh * 8);
        v16bf a = __builtin_shufflevector(alo, ahi, 0, 1, 2, 3, 4, 5, 6, 7,
                                                    8, 9, 10, 11, 12, 13, 14, 15);
#pragma unroll
        for (int t = 0; t < 4; ++t) {
            // B 32x16 fragment: lane holds column N = lane%16, K {lh*8..lh*8+7,
            // 16+lh*8..16+lh*8+7} (same K order as A).
            const __bf16* brow = cbuf + (t * 16 + lm) * BROW + lh * 8;
            v8bf blo = *(const v8bf*)brow;
            v8bf bhi = *(const v8bf*)(brow + 16);
            v16bf b = __builtin_shufflevector(blo, bhi, 0, 1, 2, 3, 4, 5, 6, 7,
                                                        8, 9, 10, 11, 12, 13, 14, 15);
            acc[t] = __builtin_amdgcn_wmma_f32_16x16x32_bf16(
                false, a, false, b, (short)0, acc[t], false, false);
        }
    }
}

// Fused fwd/rev pair of gate GEMMs (blockIdx.z selects the cell):
// G[m,n] = sum_k A1[m,k]*W1[n,k] + sum_k A2[m,k]*W2[n,k] + bias[n]
// M=256, N=2048. Grid (32, 2, 2), block 256 (8 waves); WG tile 128x64.
__global__ __launch_bounds__(256) void gates_gemm2_kernel(
    const __bf16* __restrict__ A1, int lda1, int K1,
    const __bf16* __restrict__ A2a, const __bf16* __restrict__ A2b, int lda2, int K2,
    const __bf16* __restrict__ W1a, const __bf16* __restrict__ W2a,
    const __bf16* __restrict__ W1b, const __bf16* __restrict__ W2b,
    const float* __restrict__ ba, const float* __restrict__ bb,
    float* __restrict__ Ga, float* __restrict__ Gb)
{
    __shared__ alignas(16) __bf16 bsm[2 * BTILE];   // double-buffered B tile
    const int sel = blockIdx.z;
    const __bf16* A2  = sel ? A2b : A2a;
    const __bf16* W1  = sel ? W1b : W1a;
    const __bf16* W2  = sel ? W2b : W2a;
    const float* bias = sel ? bb : ba;
    float* G          = sel ? Gb : Ga;

    const int tid  = threadIdx.x;
    const int lane = tid & 31;
    const int wave = tid >> 5;
    const int lm = lane & 15;
    const int lh = lane >> 4;
    const int n0 = blockIdx.x * 64;
    const int m0 = blockIdx.y * 128 + wave * 16;

    v8f acc[4] = {};
    mma_panel_lds(A1, lda1, K1, W1, n0, tid, lm, lh, m0 + lm, bsm, acc);
    mma_panel_lds(A2, lda2, K2, W2, n0, tid, lm, lh, m0 + lm, bsm, acc);

#pragma unroll
    for (int t = 0; t < 4; ++t) {
        const int n = n0 + t * 16 + lm;
        const float bv = bias[n];
#pragma unroll
        for (int r = 0; r < 8; ++r) {
            const int m = m0 + r + lh * 8;   // C layout: VGPR r -> M=r (+8 for hi lanes)
            G[(size_t)m * GATES + n] = acc[t][r] + bv;
        }
    }
}

// Fused fwd/rev pair of elementwise LSTM cell updates (blockIdx.y selects cell):
// c2 = sig(f)*c + sig(i)*tanh(g); h2 = sig(o)*tanh(c2)
__global__ __launch_bounds__(256) void cell_pair_kernel(
    const float* __restrict__ Ga, const float* __restrict__ Gb,
    float* __restrict__ ca, float* __restrict__ cb,
    __bf16* __restrict__ ha, __bf16* __restrict__ hb, int h_stride,
    float* __restrict__ outa, float* __restrict__ outb)
{
    const int sel = blockIdx.y;
    const float* G = sel ? Gb : Ga;
    float* c       = sel ? cb : ca;
    __bf16* h      = sel ? hb : ha;
    float* out     = sel ? outb : outa;

    const int idx = blockIdx.x * 256 + threadIdx.x;  // < BATCH*HID
    const int b = idx >> 9;
    const int j = idx & 511;
    const float* g = G + (size_t)b * GATES;
    const float gi = sigmoid_(g[j]);
    const float gf = sigmoid_(g[512 + j]);
    const float gg = tanhf(g[1024 + j]);
    const float go = sigmoid_(g[1536 + j]);
    const float c2 = gf * c[idx] + gi * gg;
    const float h2 = go * tanhf(c2);
    c[idx] = c2;
    out[idx] = c2;
    h[(size_t)b * h_stride + j] = to_bf16(h2);
}

__global__ __launch_bounds__(256) void embed_gather_kernel(
    const int* __restrict__ tok, const float* __restrict__ embed,
    __bf16* __restrict__ x)
{
    const int idx = blockIdx.x * 256 + threadIdx.x;  // < S*B*E
    const int row = idx / EMB;
    const int col = idx - row * EMB;
    const int t = tok[row];
    x[idx] = to_bf16(embed[(size_t)t * EMB + col]);
}

__global__ __launch_bounds__(256) void cvt_bf16_kernel(
    const float* __restrict__ s, __bf16* __restrict__ d, int n)
{
    const int i = blockIdx.x * 256 + threadIdx.x;
    if (i < n) d[i] = to_bf16(s[i]);
}

__global__ __launch_bounds__(256) void bias_sum_kernel(
    const float* __restrict__ a, const float* __restrict__ b,
    float* __restrict__ d, int n)
{
    const int i = blockIdx.x * 256 + threadIdx.x;
    if (i < n) d[i] = a[i] + b[i];
}

__global__ __launch_bounds__(256) void zero_u32_kernel(unsigned* __restrict__ p, int n)
{
    const int i = blockIdx.x * 256 + threadIdx.x;
    if (i < n) p[i] = 0u;
}

extern "C" void kernel_launch(void* const* d_in, const int* in_sizes, int n_in,
                              void* d_out, int out_size, void* d_ws, size_t ws_size,
                              hipStream_t stream) {
    const int*   tok   = (const int*)d_in[0];
    const float* embed = (const float*)d_in[1];
    // Cell order: 0=l0f, 1=l0r, 2=l1f, 3=l1r; inputs at 2 + 4*cell.
    const float* w_ih[4]; const float* w_hh[4];
    const float* b_ih[4]; const float* b_hh[4];
    const int Kih[4] = {EMB, EMB, 2 * HID, 2 * HID};
    for (int c = 0; c < 4; ++c) {
        w_ih[c] = (const float*)d_in[2 + 4 * c + 0];
        w_hh[c] = (const float*)d_in[2 + 4 * c + 1];
        b_ih[c] = (const float*)d_in[2 + 4 * c + 2];
        b_hh[c] = (const float*)d_in[2 + 4 * c + 3];
    }
    float* out = (float*)d_out;

    // ---- workspace bump allocator (256B aligned) ----
    char* ws = (char*)d_ws;
    size_t off = 0;
    auto alloc = [&](size_t bytes) -> void* {
        void* p = ws + off;
        off = (off + bytes + 255) & ~(size_t)255;
        return p;
    };

    __bf16* x_bf = (__bf16*)alloc((size_t)S_LEN * BATCH * EMB * 2);
    __bf16* wih_bf[4]; __bf16* whh_bf[4]; float* bias[4]; float* G[4];
    for (int c = 0; c < 4; ++c) {
        wih_bf[c] = (__bf16*)alloc((size_t)GATES * Kih[c] * 2);
        whh_bf[c] = (__bf16*)alloc((size_t)GATES * HID * 2);
        bias[c]   = (float*)alloc((size_t)GATES * 4);
        G[c]      = (float*)alloc((size_t)BATCH * GATES * 4);
    }
    // States, contiguous so one zero kernel covers all.
    const size_t state_bytes = (size_t)BATCH * 1024 * 2        // hcat (y0) bf16
                             + 2 * (size_t)BATCH * HID * 2     // h1f, h1b bf16
                             + 4 * (size_t)BATCH * HID * 4;    // c[4] fp32
    char* st = (char*)alloc(state_bytes);
    __bf16* hcat = (__bf16*)st;
    __bf16* h1f  = (__bf16*)(st + (size_t)BATCH * 1024 * 2);
    __bf16* h1b  = h1f + (size_t)BATCH * HID;
    float*  cst  = (float*)(st + (size_t)BATCH * 1024 * 2 + 2 * (size_t)BATCH * HID * 2);

    // ---- setup ----
    for (int c = 0; c < 4; ++c) {
        int nih = GATES * Kih[c];
        cvt_bf16_kernel<<<(nih + 255) / 256, 256, 0, stream>>>(w_ih[c], wih_bf[c], nih);
        int nhh = GATES * HID;
        cvt_bf16_kernel<<<(nhh + 255) / 256, 256, 0, stream>>>(w_hh[c], whh_bf[c], nhh);
        bias_sum_kernel<<<(GATES + 255) / 256, 256, 0, stream>>>(b_ih[c], b_hh[c], bias[c], GATES);
    }
    {
        int n = S_LEN * BATCH * EMB;
        embed_gather_kernel<<<(n + 255) / 256, 256, 0, stream>>>(tok, embed, x_bf);
        int nz = (int)(state_bytes / 4);
        zero_u32_kernel<<<(nz + 255) / 256, 256, 0, stream>>>((unsigned*)st, nz);
    }

    // ---- recurrence: 4 launches per timestep ----
    const dim3 ggrid(GATES / 64, BATCH / 128, 2);
    const dim3 cgrid(BH / 256, 2);
    for (int t = 0; t < S_LEN; ++t) {
        const __bf16* xt = x_bf + (size_t)t * BATCH * EMB;
        float* out_t = out + (size_t)t * 4 * BH;
        // layer 0 (reads old hcat)
        gates_gemm2_kernel<<<ggrid, 256, 0, stream>>>(
            xt, EMB, EMB, hcat, hcat + HID, 1024, HID,
            wih_bf[0], whh_bf[0], wih_bf[1], whh_bf[1],
            bias[0], bias[1], G[0], G[1]);
        cell_pair_kernel<<<cgrid, 256, 0, stream>>>(
            G[0], G[1], cst + 0 * BH, cst + 1 * BH,
            hcat, hcat + HID, 1024, out_t + 0 * BH, out_t + 1 * BH);
        // layer 1 (reads new hcat = y0, old h1f/h1b)
        gates_gemm2_kernel<<<ggrid, 256, 0, stream>>>(
            hcat, 1024, 2 * HID, h1f, h1b, HID, HID,
            wih_bf[2], whh_bf[2], wih_bf[3], whh_bf[3],
            bias[2], bias[3], G[2], G[3]);
        cell_pair_kernel<<<cgrid, 256, 0, stream>>>(
            G[2], G[3], cst + 2 * BH, cst + 3 * BH,
            h1f, h1b, HID, out_t + 2 * BH, out_t + 3 * BH);
    }
}
